// MaxPoolAggregator_43593918054684
// MI455X (gfx1250) — matrix operand order repaired and checked by reading, not compiled, not run on hardware
//
#include <hip/hip_runtime.h>
#include <hip/hip_bf16.h>

typedef __attribute__((ext_vector_type(16))) _Float16 v16h;
typedef __attribute__((ext_vector_type(8)))  float    v8f;

#define D 128  // D_IN == D_OUT == 128

// Order-preserving float->uint encoding: a <= b  <=>  enc(a) <= enc(b) (as uint)
__device__ __forceinline__ unsigned int enc_f32(float f) {
    unsigned int u = __float_as_uint(f);
    return (u & 0x80000000u) ? ~u : (u | 0x80000000u);
}
__device__ __forceinline__ float dec_f32(unsigned int u) {
    return __uint_as_float((u & 0x80000000u) ? (u & 0x7fffffffu) : ~u);
}

// ---------------------------------------------------------------------------
// Kernel 1: out[:, 0:128] = x ; out[:, 128:256] (as uint) = 0 ; deg = 0
// ---------------------------------------------------------------------------
__global__ void mpagg_init_kernel(const float* __restrict__ x,
                                  float* __restrict__ out,
                                  int* __restrict__ deg, int n_nodes) {
    long long tid = (long long)blockIdx.x * blockDim.x + threadIdx.x;
    long long total = (long long)n_nodes * D;
    if (tid >= total) return;
    int n = (int)(tid / D);
    int c = (int)(tid % D);
    out[(long long)n * (2 * D) + c] = x[tid];
    ((unsigned int*)out)[(long long)n * (2 * D) + D + c] = 0u;  // enc identity (< enc(-inf))
    if (c == 0) deg[n] = 0;
}

// ---------------------------------------------------------------------------
// Kernel 2: norm_m = x @ W1 via v_wmma_f32_16x16x32_f16, stored ENCODED to ws.
// One wave32 per 16x16 output tile; block = 8 waves = the 8 column tiles of a
// 16-row strip. K = 128 -> 4 WMMA steps. 50000 rows = 3125 * 16 (no tail).
// ---------------------------------------------------------------------------
__global__ void mpagg_gemm_wmma_kernel(const float* __restrict__ x,
                                       const float* __restrict__ W1,
                                       unsigned int* __restrict__ nm_enc) {
    const int lane = threadIdx.x & 31;
    const int wave = threadIdx.x >> 5;       // 0..7 -> column tile
    const int h    = lane >> 4;              // lane half
    const int r    = lane & 15;              // row (A) / column (B,C,D) index
    const int row0 = blockIdx.x * 16;
    const int col0 = wave * 16;

    v8f acc = {};
    #pragma unroll
    for (int kb = 0; kb < 4; ++kb) {
        const int k0 = kb * 32;
        v16h a, b;
        // A fragment (16x32 f16): lane half h, vgpr v holds K = 2v + (v>=4?8:0) + 8h, +1
        const float* arow = x + (long long)(row0 + r) * D + k0;
        #pragma unroll
        for (int v = 0; v < 8; ++v) {
            int K = 2 * v + ((v >= 4) ? 8 : 0) + 8 * h;
            a[2 * v]     = (_Float16)arow[K];
            a[2 * v + 1] = (_Float16)arow[K + 1];
        }
        // B fragment (32x16 f16): column n = r, vgpr v holds K = 2v + (v>=4?8:0) + 8h, +1
        const float* bcol = W1 + (long long)k0 * D + (col0 + r);
        #pragma unroll
        for (int v = 0; v < 8; ++v) {
            int K = 2 * v + ((v >= 4) ? 8 : 0) + 8 * h;
            b[2 * v]     = (_Float16)bcol[(long long)K * D];
            b[2 * v + 1] = (_Float16)bcol[(long long)(K + 1) * D];
        }
        acc = __builtin_amdgcn_wmma_f32_16x16x32_f16(
            /*neg_a=*/false, a, /*neg_b=*/false, b,
            /*c_mod=*/(short)0, acc, /*reuse_a=*/false, /*reuse_b=*/false);
    }
    // D fragment: element (M = v + 8h, N = r); store order-encoded for atomics.
    unsigned int* orow = nm_enc + (long long)row0 * D + col0 + r;
    #pragma unroll
    for (int v = 0; v < 8; ++v) {
        int M = v + 8 * h;
        orow[(long long)M * D] = enc_f32(acc[v]);
    }
}

// ---------------------------------------------------------------------------
// Kernel 3: one wave per edge. Gather encoded row of src (uint4 per lane,
// 512B/edge, L2-resident), scatter-max into encoded pooled region of out.
// ---------------------------------------------------------------------------
__global__ void mpagg_edge_kernel(const int* __restrict__ erow,
                                  const int* __restrict__ ecol,
                                  const unsigned int* __restrict__ nm_enc,
                                  unsigned int* __restrict__ out_u,
                                  int* __restrict__ deg, int n_edges) {
    int e = blockIdx.x * (blockDim.x >> 5) + (threadIdx.x >> 5);
    if (e >= n_edges) return;
    const int lane = threadIdx.x & 31;
    const int src = erow[e];
    const int dst = ecol[e];
    const uint4 v = ((const uint4*)(nm_enc + (long long)src * D))[lane];
    unsigned int* ob = out_u + (long long)dst * (2 * D) + D + lane * 4;
    atomicMax(ob + 0, v.x);
    atomicMax(ob + 1, v.y);
    atomicMax(ob + 2, v.z);
    atomicMax(ob + 3, v.w);
    if (lane == 0) atomicAdd(deg + dst, 1);
}

// ---------------------------------------------------------------------------
// Kernel 4: decode pooled region in place; zero rows with deg == 0.
// ---------------------------------------------------------------------------
__global__ void mpagg_finalize_kernel(float* __restrict__ out,
                                      const int* __restrict__ deg, int n_nodes) {
    long long tid = (long long)blockIdx.x * blockDim.x + threadIdx.x;
    long long total = (long long)n_nodes * D;
    if (tid >= total) return;
    int n = (int)(tid / D);
    int c = (int)(tid % D);
    long long idx = (long long)n * (2 * D) + D + c;
    unsigned int u = ((unsigned int*)out)[idx];
    float f = (deg[n] > 0) ? dec_f32(u) : 0.0f;
    out[idx] = f;
}

extern "C" void kernel_launch(void* const* d_in, const int* in_sizes, int n_in,
                              void* d_out, int out_size, void* d_ws, size_t ws_size,
                              hipStream_t stream) {
    const float* x   = (const float*)d_in[0];
    const float* W1  = (const float*)d_in[1];
    const int*   ei  = (const int*)d_in[2];     // [2, E] int32
    const int n_nodes = in_sizes[0] / D;        // 50000
    const int n_edges = in_sizes[2] / 2;        // 800000
    const int* erow = ei;                       // edge_index[0] = source (gather)
    const int* ecol = ei + n_edges;             // edge_index[1] = destination (scatter)

    float* out = (float*)d_out;
    unsigned int* nm_enc = (unsigned int*)d_ws;                       // 50000*128*4 = 25.6 MB
    int* deg = (int*)((char*)d_ws + (size_t)n_nodes * D * sizeof(unsigned int));

    const int threads = 256;
    long long elems = (long long)n_nodes * D;
    int blocks_elem = (int)((elems + threads - 1) / threads);

    // 1) copy x into out, init encoded pooled region + deg
    mpagg_init_kernel<<<blocks_elem, threads, 0, stream>>>(x, out, deg, n_nodes);

    // 2) WMMA GEMM -> encoded norm_m in workspace (3125 row strips, 8 waves/block)
    mpagg_gemm_wmma_kernel<<<n_nodes / 16, threads, 0, stream>>>(x, W1, nm_enc);

    // 3) edge scatter-max (one wave32 per edge, 8 edges per block)
    int edge_blocks = (n_edges + 7) / 8;
    mpagg_edge_kernel<<<edge_blocks, threads, 0, stream>>>(erow, ecol, nm_enc,
                                                           (unsigned int*)d_out, deg, n_edges);

    // 4) decode + zero empty segments
    mpagg_finalize_kernel<<<blocks_elem, threads, 0, stream>>>(out, deg, n_nodes);
}
